// EnAttention_49555332662092
// MI455X (gfx1250) — compile-verified, matrix operand order checked by reading, not run on hardware
//
#include <hip/hip_runtime.h>
#include <hip/hip_bf16.h>

// ---------------- types / helpers ----------------
typedef unsigned short     u16;
typedef unsigned int       u32;
typedef unsigned long long u64;
typedef __attribute__((ext_vector_type(16))) __bf16   v16bf;
typedef __attribute__((ext_vector_type(8)))  float    v8f;
typedef __attribute__((ext_vector_type(4)))  unsigned u32x4;

#define Nn 50000
#define Ee 640000

static const size_t OFF_HBF   = 0;                      // 50000*128*2 = 12,800,000
static const size_t OFF_WQKVT = 12800000;               // 1536*128*2  =    393,216
static const size_t OFF_WOUTT = OFF_WQKVT + 393216;     // 128*512*2   =    131,072
static const size_t OFF_WE1T  = OFF_WOUTT + 131072;     // 128*256*2   =     65,536
static const size_t OFF_WE2T  = OFF_WE1T  + 65536;      // 128*128*2   =     32,768
static const size_t OFF_WC1T  = OFF_WE2T  + 32768;      // 128*128*2   =     32,768
static const size_t OFF_EW    = OFF_WC1T  + 32768;      // 640000*4    =  2,560,000
static const size_t OFF_RED   = OFF_EW    + 2560000;    // 16 bytes

__device__ __forceinline__ u16 f2bf(float f) {
  __hip_bfloat16 b = __float2bfloat16(f);   // hardware v_cvt_*bf16_f32 path (RNE)
  u16 u; __builtin_memcpy(&u, &b, sizeof(u));
  return u;
}
__device__ __forceinline__ float bf2f(u16 h) { return __uint_as_float(((u32)h) << 16); }
__device__ __forceinline__ float silu(float v) { return v / (1.0f + __expf(-v)); }

__device__ __forceinline__ u32 encf(float f) {
  u32 u = __float_as_uint(f);
  return (u & 0x80000000u) ? ~u : (u | 0x80000000u);
}
__device__ __forceinline__ float decf(u32 u) {
  return (u & 0x80000000u) ? __uint_as_float(u & 0x7FFFFFFFu) : __uint_as_float(~u);
}

// ---- CDNA5 async global->LDS copies (ASYNCcnt path, no VGPR round-trip) ----
// Per-lane: copies global data at vaddr+off into LDS at vdst+off; EXEC all-ones here.
__device__ __forceinline__ void async_row_b64(u16* lds, const u16* g, int lane) {
  u32 la = (u32)(size_t)lds + lane * 8;              // 32 lanes x 8B = 256B row
  u64 ga = (u64)(size_t)g + lane * 8;
  asm volatile("global_load_async_to_lds_b64 %0, %1, off" :: "v"(la), "v"(ga) : "memory");
}
__device__ __forceinline__ void async_blk_b128(u16* lds, const u16* g, int lane) {
  u32 la = (u32)(size_t)lds + lane * 16;             // 32 lanes x 16B = 512B block
  u64 ga = (u64)(size_t)g + lane * 16;
  asm volatile("global_load_async_to_lds_b128 %0, %1, off" :: "v"(la), "v"(ga) : "memory");
}
__device__ __forceinline__ void wait_async0() {
  asm volatile("s_wait_asynccnt 0x0" ::: "memory");
}

// Load a 16x32 bf16 A/B fragment (ISA 7.12.2 layout): lane l holds row
// (rowBase + l&15); K elements at kBase + g*8..+7 and kBase+16+g*8..+7 (g=l>>4).
__device__ __forceinline__ v16bf ld_frag(const u16* base, int stride, int rowBase,
                                         int kBase, int lane) {
  const int g = lane >> 4;
  const u16* rp = base + (size_t)(rowBase + (lane & 15)) * stride + kBase + g * 8;
  union { v16bf v; u32x4 q[2]; } f;
  f.q[0] = *(const u32x4*)(rp);
  f.q[1] = *(const u32x4*)(rp + 16);
  return f.v;
}

#define WMMA_BF16(acc, a, b) \
  __builtin_amdgcn_wmma_f32_16x16x32_bf16(false, (a), false, (b), (short)0, (acc), false, false)

// ---------------- prep kernels ----------------
__global__ void k_h2bf(const float* __restrict__ src, u16* __restrict__ dst, int n) {
  int i = blockIdx.x * blockDim.x + threadIdx.x;
  if (i < n) dst[i] = f2bf(src[i]);
}

// src is (I x O) row-major f32; dst is (O x I) row-major bf16 (transposed)
__global__ void k_transpose(const float* __restrict__ src, u16* __restrict__ dst,
                            int I, int O) {
  int idx = blockIdx.x * blockDim.x + threadIdx.x;
  if (idx >= I * O) return;
  int o = idx / I, i = idx % I;
  dst[idx] = f2bf(src[(size_t)i * O + o]);
}

__global__ void k_zero(float* __restrict__ p, int n) {
  int i = blockIdx.x * blockDim.x + threadIdx.x;
  if (i < n) p[i] = 0.0f;
}

// ---------------- node branch: qkv -> attn(8x8) -> out ----------------
__global__ void __launch_bounds__(32) k_node(const u16* __restrict__ hbf,
                                             const u16* __restrict__ WqkvT,
                                             const u16* __restrict__ WoutT,
                                             const float* __restrict__ bqkv,
                                             const float* __restrict__ bout,
                                             float* __restrict__ out) {
  __shared__ u16  hA[16 * 128];       //  4 KB : A tile (16 nodes x 128) bf16
  __shared__ u32  region[8192];       // 32 KB : qk bf16 (16x1024) then feats f32 (16x512)
  __shared__ float attn[16 * 64];     //  4 KB : dots/attn [m][h*8+j]
  __shared__ u16  featsbf[16 * 512];  // 16 KB : feats bf16 for out GEMM

  const int lane = threadIdx.x;
  const int g = lane >> 4, nl = lane & 15;
  const int m0 = blockIdx.x * 16;

  // stage h rows (bf16, 4KB contiguous) via async DMA
#pragma unroll
  for (int i = 0; i < 8; ++i)
    async_blk_b128(hA + i * 256, hbf + (size_t)m0 * 128 + i * 256, lane);
  wait_async0();
  __syncthreads();

  // ---- q,k GEMM: 16x128 @ 128x1024 -> bf16 in region ----
  u16* qk = (u16*)region;  // stride 1024
  for (int nt = 0; nt < 64; ++nt) {
    v8f acc = {};
#pragma unroll
    for (int kc = 0; kc < 4; ++kc) {
      v16bf a = ld_frag(hA, 128, 0, kc * 32, lane);
      v16bf b = ld_frag(WqkvT, 128, nt * 16, kc * 32, lane);
      acc = WMMA_BF16(acc, a, b);
    }
    int c = nt * 16 + nl;
    float bb = bqkv[c];
#pragma unroll
    for (int j = 0; j < 8; ++j) qk[(j + 8 * g) * 1024 + c] = f2bf(acc[j] + bb);
  }
  __syncthreads();

  // ---- dots[m][h][j] = (q_h . k_j) / 8 ----
  for (int t = 0; t < 32; ++t) {
    int hj = g * 32 + t, hh = hj >> 3, jj = hj & 7, m = nl;
    const u32* qp = (const u32*)(qk + m * 1024 + hh * 64);
    const u32* kp = (const u32*)(qk + m * 1024 + 512 + jj * 64);
    float s = 0.f;
#pragma unroll
    for (int d = 0; d < 32; ++d) {
      u32 qa = qp[d], ka = kp[d];
      s += bf2f((u16)qa) * bf2f((u16)ka);
      s += bf2f((u16)(qa >> 16)) * bf2f((u16)(ka >> 16));
    }
    attn[m * 64 + hj] = s * 0.125f;
  }
  __syncthreads();

  // ---- softmax over j (rows of 8); 4 rows per lane ----
  for (int t = 0; t < 4; ++t) {
    float* row = attn + (lane * 4 + t) * 8;
    float mx = row[0];
#pragma unroll
    for (int j = 1; j < 8; ++j) mx = fmaxf(mx, row[j]);
    float e[8], ss = 0.f;
#pragma unroll
    for (int j = 0; j < 8; ++j) { e[j] = __expf(row[j] - mx); ss += e[j]; }
    float inv = 1.f / ss;
#pragma unroll
    for (int j = 0; j < 8; ++j) row[j] = e[j] * inv;
  }
  __syncthreads();

  // ---- v GEMM tiles, folded through attn into feats (f32 LDS, reuses region) ----
  float* featsf = (float*)region;  // 16 x 512
  for (int i = lane; i < 16 * 512; i += 32) featsf[i] = 0.f;
  __syncthreads();
  for (int nt = 0; nt < 32; ++nt) {
    v8f acc = {};
#pragma unroll
    for (int kc = 0; kc < 4; ++kc) {
      v16bf a = ld_frag(hA, 128, 0, kc * 32, lane);
      v16bf b = ld_frag(WqkvT, 128, 1024 + nt * 16, kc * 32, lane);
      acc = WMMA_BF16(acc, a, b);
    }
    int vcol = nt * 16 + nl;
    float bb = bqkv[1024 + vcol];
    int jh = nt >> 2, dcol = vcol & 63;
#pragma unroll
    for (int j = 0; j < 8; ++j) {
      int m = j + 8 * g;
      float val = acc[j] + bb;
      const float* am = attn + m * 64;
      float* fm = featsf + m * 512 + dcol;
#pragma unroll
      for (int hh = 0; hh < 8; ++hh) fm[hh * 64] += am[hh * 8 + jh] * val;
    }
  }
  __syncthreads();

  for (int i = lane; i < 16 * 512; i += 32) featsbf[i] = f2bf(featsf[i]);
  __syncthreads();

  // ---- out projection: 16x512 @ 512x128 ----
  for (int nt = 0; nt < 8; ++nt) {
    v8f acc = {};
#pragma unroll
    for (int kc = 0; kc < 16; ++kc) {
      v16bf a = ld_frag(featsbf, 512, 0, kc * 32, lane);
      v16bf b = ld_frag(WoutT, 512, nt * 16, kc * 32, lane);
      acc = WMMA_BF16(acc, a, b);
    }
    int c = nt * 16 + nl;
    float bb = bout[c];
#pragma unroll
    for (int j = 0; j < 8; ++j) out[(size_t)(m0 + j + 8 * g) * 128 + c] = acc[j] + bb;
  }
}

// ---------------- edge branch: gather -> MLPs -> ew[e] ----------------
// One wave per block, 4 M-tiles (64 edges): each B fragment is loaded once and
// reused by 4 accumulators -> 4x less weight traffic from L2.
__global__ void __launch_bounds__(32) k_edge(const u16* __restrict__ hbf,
                                             const float* __restrict__ x,
                                             const int* __restrict__ ei,
                                             const u16* __restrict__ We1T,
                                             const u16* __restrict__ We2T,
                                             const u16* __restrict__ Wc1T,
                                             const float* __restrict__ We1f,
                                             const float* __restrict__ be1,
                                             const float* __restrict__ be2,
                                             const float* __restrict__ bc1,
                                             const float* __restrict__ Wc2f,
                                             const float* __restrict__ bc2,
                                             float* __restrict__ ew) {
  __shared__ u16  sh[4][16][128];  // h[row] bf16, later edge_features
  __shared__ u16  sc[4][16][128];  // h[col] bf16, later ef3
  __shared__ u16  st[4][16][128];  // stage (ef1)
  __shared__ float sd[64];         // rel_dist

  const int lane = threadIdx.x;
  const int g = lane >> 4, nl = lane & 15;
  const int e0 = blockIdx.x * 64;

  // async-DMA gather of 128 bf16 rows (256B each) into LDS
  for (int t = 0; t < 4; ++t) {
#pragma unroll
    for (int r = 0; r < 16; ++r) {
      int er = e0 + t * 16 + r;
      int ir = ei[er], ic = ei[Ee + er];
      async_row_b64(&sh[t][r][0], hbf + (size_t)ir * 128, lane);
      async_row_b64(&sc[t][r][0], hbf + (size_t)ic * 128, lane);
    }
  }
  // rel_dist for 64 edges (2 per lane), overlapped with async copies
#pragma unroll
  for (int t2 = 0; t2 < 2; ++t2) {
    int idx = t2 * 32 + lane;
    int er = e0 + idx;
    int ir = ei[er], ic = ei[Ee + er];
    float dx = x[3 * ir + 0] - x[3 * ic + 0];
    float dy = x[3 * ir + 1] - x[3 * ic + 1];
    float dz = x[3 * ir + 2] - x[3 * ic + 2];
    sd[idx] = dx * dx + dy * dy + dz * dz;
  }
  wait_async0();
  __syncthreads();

  // GEMM1: [h_row | h_col] (16x256) @ We1[0:256] + rel_dist*We1[256] + be1, SiLU -> st
  for (int nt = 0; nt < 8; ++nt) {
    v8f acc[4] = {{}, {}, {}, {}};
#pragma unroll
    for (int kc = 0; kc < 8; ++kc) {
      v16bf b = ld_frag(We1T, 256, nt * 16, kc * 32, lane);
      const u16* ab = (kc < 4) ? &sh[0][0][0] : &sc[0][0][0];
      int kb = (kc & 3) * 32;
#pragma unroll
      for (int t = 0; t < 4; ++t)
        acc[t] = WMMA_BF16(acc[t], ld_frag(ab + t * 16 * 128, 128, 0, kb, lane), b);
    }
    int c = nt * 16 + nl;
    float b1 = be1[c], w256 = We1f[256 * 128 + c];
#pragma unroll
    for (int t = 0; t < 4; ++t)
#pragma unroll
      for (int j = 0; j < 8; ++j) {
        int m = j + 8 * g;
        st[t][m][c] = f2bf(silu(acc[t][j] + b1 + sd[t * 16 + m] * w256));
      }
  }
  __syncthreads();

  // GEMM2: ef1 @ We2 + be2, SiLU -> sh (edge_features)
  for (int nt = 0; nt < 8; ++nt) {
    v8f acc[4] = {{}, {}, {}, {}};
#pragma unroll
    for (int kc = 0; kc < 4; ++kc) {
      v16bf b = ld_frag(We2T, 128, nt * 16, kc * 32, lane);
#pragma unroll
      for (int t = 0; t < 4; ++t)
        acc[t] = WMMA_BF16(acc[t], ld_frag(&st[t][0][0], 128, 0, kc * 32, lane), b);
    }
    int c = nt * 16 + nl;
    float b2 = be2[c];
#pragma unroll
    for (int t = 0; t < 4; ++t)
#pragma unroll
      for (int j = 0; j < 8; ++j) sh[t][j + 8 * g][c] = f2bf(silu(acc[t][j] + b2));
  }
  __syncthreads();

  // GEMM3: edge_features @ Wc1 + bc1, SiLU -> sc
  for (int nt = 0; nt < 8; ++nt) {
    v8f acc[4] = {{}, {}, {}, {}};
#pragma unroll
    for (int kc = 0; kc < 4; ++kc) {
      v16bf b = ld_frag(Wc1T, 128, nt * 16, kc * 32, lane);
#pragma unroll
      for (int t = 0; t < 4; ++t)
        acc[t] = WMMA_BF16(acc[t], ld_frag(&sh[t][0][0], 128, 0, kc * 32, lane), b);
    }
    int c = nt * 16 + nl;
    float b3 = bc1[c];
#pragma unroll
    for (int t = 0; t < 4; ++t)
#pragma unroll
      for (int j = 0; j < 8; ++j) sc[t][j + 8 * g][c] = f2bf(silu(acc[t][j] + b3));
  }
  __syncthreads();

  // dot with Wc2 (128x1) -> ew (4 tiles)
  for (int t = 0; t < 4; ++t) {
    float s = 0.f;
    for (int c = g * 64; c < g * 64 + 64; ++c) s += bf2f(sc[t][nl][c]) * Wc2f[c];
    s += __shfl_down(s, 16);
    if (lane < 16) ew[e0 + t * 16 + lane] = s + bc2[0];
  }
}

// ---------------- global softmax over edges + scatter ----------------
__global__ void k_max(const float* __restrict__ ew, u32* __restrict__ rmax, int n) {
  __shared__ float sm[256];
  float m = -3.402823466e38f;
  for (int i = blockIdx.x * blockDim.x + threadIdx.x; i < n; i += gridDim.x * blockDim.x)
    m = fmaxf(m, ew[i]);
  sm[threadIdx.x] = m;
  __syncthreads();
  for (int s = 128; s > 0; s >>= 1) {
    if (threadIdx.x < s) sm[threadIdx.x] = fmaxf(sm[threadIdx.x], sm[threadIdx.x + s]);
    __syncthreads();
  }
  if (threadIdx.x == 0) atomicMax(rmax, encf(sm[0]));
}

__global__ void k_sum(const float* __restrict__ ew, const u32* __restrict__ rmax,
                      float* __restrict__ rsum, int n) {
  __shared__ float sm[256];
  float mv = decf(*rmax);
  float a = 0.f;
  for (int i = blockIdx.x * blockDim.x + threadIdx.x; i < n; i += gridDim.x * blockDim.x)
    a += __expf(ew[i] - mv);
  sm[threadIdx.x] = a;
  __syncthreads();
  for (int s = 128; s > 0; s >>= 1) {
    if (threadIdx.x < s) sm[threadIdx.x] += sm[threadIdx.x + s];
    __syncthreads();
  }
  if (threadIdx.x == 0) atomicAdd(rsum, sm[0]);
}

__global__ void k_scatter(const float* __restrict__ ew, const int* __restrict__ ei,
                          const float* __restrict__ x, const u32* __restrict__ rmax,
                          const float* __restrict__ rsum, float* __restrict__ coor) {
  int e = blockIdx.x * blockDim.x + threadIdx.x;
  if (e >= Ee) return;
  float mv = decf(*rmax);
  float wgt = __expf(ew[e] - mv) * (1.f / rsum[0]);
  int r = ei[e], c = ei[Ee + e];
  float rx = x[3 * r + 0] - x[3 * c + 0];
  float ry = x[3 * r + 1] - x[3 * c + 1];
  float rz = x[3 * r + 2] - x[3 * c + 2];
  atomicAdd(&coor[3 * r + 0], wgt * rx);
  atomicAdd(&coor[3 * r + 1], wgt * ry);
  atomicAdd(&coor[3 * r + 2], wgt * rz);
}

// ---------------- launcher ----------------
extern "C" void kernel_launch(void* const* d_in, const int* in_sizes, int n_in,
                              void* d_out, int out_size, void* d_ws, size_t ws_size,
                              hipStream_t stream) {
  (void)in_sizes; (void)n_in; (void)out_size; (void)ws_size;
  const float* h    = (const float*)d_in[0];
  const float* x    = (const float*)d_in[1];
  const int*   ei   = (const int*)d_in[2];
  const float* Wqkv = (const float*)d_in[3];
  const float* bqkv = (const float*)d_in[4];
  const float* Wout = (const float*)d_in[5];
  const float* bout = (const float*)d_in[6];
  const float* We1  = (const float*)d_in[7];
  const float* be1  = (const float*)d_in[8];
  const float* We2  = (const float*)d_in[9];
  const float* be2  = (const float*)d_in[10];
  const float* Wc1  = (const float*)d_in[11];
  const float* bc1  = (const float*)d_in[12];
  const float* Wc2  = (const float*)d_in[13];
  const float* bc2  = (const float*)d_in[14];

  char* ws = (char*)d_ws;
  u16* hbf    = (u16*)(ws + OFF_HBF);
  u16* WqkvT  = (u16*)(ws + OFF_WQKVT);
  u16* WoutT  = (u16*)(ws + OFF_WOUTT);
  u16* We1T   = (u16*)(ws + OFF_WE1T);
  u16* We2T   = (u16*)(ws + OFF_WE2T);
  u16* Wc1T   = (u16*)(ws + OFF_WC1T);
  float* ewb  = (float*)(ws + OFF_EW);
  u32* rmax   = (u32*)(ws + OFF_RED);
  float* rsum = (float*)(ws + OFF_RED + 4);

  float* feats = (float*)d_out;
  float* coor  = (float*)d_out + (size_t)Nn * 128;

  // prep: h -> bf16, weights -> bf16 transposed
  k_h2bf<<<(Nn * 128 + 255) / 256, 256, 0, stream>>>(h, hbf, Nn * 128);
  k_transpose<<<(128 * 1536 + 255) / 256, 256, 0, stream>>>(Wqkv, WqkvT, 128, 1536);
  k_transpose<<<(512 * 128 + 255) / 256, 256, 0, stream>>>(Wout, WoutT, 512, 128);
  k_transpose<<<(256 * 128 + 255) / 256, 256, 0, stream>>>(We1, We1T, 256, 128);
  k_transpose<<<(128 * 128 + 255) / 256, 256, 0, stream>>>(We2, We2T, 128, 128);
  k_transpose<<<(128 * 128 + 255) / 256, 256, 0, stream>>>(Wc1, Wc1T, 128, 128);

  // init outputs / reduction cells
  k_zero<<<(Nn * 3 + 255) / 256, 256, 0, stream>>>(coor, Nn * 3);
  k_zero<<<1, 32, 0, stream>>>((float*)(ws + OFF_RED), 4);

  // node branch (N/16 = 3125 tiles, one wave each)
  k_node<<<Nn / 16, 32, 0, stream>>>(hbf, WqkvT, WoutT, bqkv, bout, feats);

  // edge branch (E/64 = 10000 blocks, one wave handling 4 tiles)
  k_edge<<<Ee / 64, 32, 0, stream>>>(hbf, x, ei, We1T, We2T, Wc1T,
                                     We1, be1, be2, bc1, Wc2, bc2, ewb);

  // global softmax over edges + scatter-add of weighted rel_pos
  k_max<<<512, 256, 0, stream>>>(ewb, rmax, Ee);
  k_sum<<<512, 256, 0, stream>>>(ewb, rmax, rsum, Ee);
  k_scatter<<<Ee / 256, 256, 0, stream>>>(ewb, ei, x, rmax, rsum, coor);
}
